// Model_58342835749130
// MI455X (gfx1250) — compile-verified, matrix operand order checked by reading, not run on hardware
//
#include <hip/hip_runtime.h>

#define B_Q     1024
#define N_CAND  200000
#define F_IN    32
#define D_MODEL 128
#define D_BLK   256
#define C_CTX   96

typedef unsigned short u16;
typedef __attribute__((ext_vector_type(16))) __bf16 bf16x16;
typedef __attribute__((ext_vector_type(16))) u16    u16x16;
typedef __attribute__((ext_vector_type(8)))  u16    u16x8;
typedef __attribute__((ext_vector_type(8)))  float  v8f;

static __device__ __forceinline__ u16 f2bf(float f) {
  unsigned u = __float_as_uint(f);
  u += 0x7FFFu + ((u >> 16) & 1u);   // round-to-nearest-even
  return (u16)(u >> 16);
}
static __device__ __forceinline__ float bf2f(u16 h) {
  return __uint_as_float(((unsigned)h) << 16);
}

// Load one 16x32 bf16 WMMA operand fragment.
// Works for A from row-major [M][K] and for B when the source is stored
// [N][K] (transposed weights / candidate-major keys).
// ISA 16-bit A layout: lanes 0-15 row m, VGPR0-3 = K g*8+0..7, VGPR4-7 = K g*8+16..23.
static __device__ __forceinline__ u16x16 load_frag(const u16* p, int ld, int k0) {
  int lane = threadIdx.x & 31;
  int m = lane & 15, g = lane >> 4;
  const u16* row = p + (size_t)m * ld + k0 + g * 8;
  u16x8 lo = *(const u16x8*)(row);
  u16x8 hi = *(const u16x8*)(row + 16);
  u16x16 r;
#pragma unroll
  for (int i = 0; i < 8; ++i) { r[i] = lo[i]; r[8 + i] = hi[i]; }
  return r;
}

static __device__ __forceinline__ v8f wmma_bf16(u16x16 a, u16x16 b, v8f c) {
  return __builtin_amdgcn_wmma_f32_16x16x32_bf16(
      false, __builtin_bit_cast(bf16x16, a),
      false, __builtin_bit_cast(bf16x16, b),
      (short)0, c, false, false);
}

// ---------------------------------------------------------------------------
// Weight prep: f32 [in][out]  ->  bf16 [out][in]  (B-fragment friendly layout)
// ---------------------------------------------------------------------------
__global__ void __launch_bounds__(256) transpose_cvt_kernel(
    const float* __restrict__ src, u16* __restrict__ dst, int in_dim, int out_dim) {
  int i = blockIdx.x * 256 + threadIdx.x;
  if (i >= in_dim * out_dim) return;
  int o = i % out_dim, in = i / out_dim;
  dst[(size_t)o * in_dim + in] = f2bf(src[i]);
}

// ---------------------------------------------------------------------------
// Encoder: one wave (32 threads) processes 16 rows end-to-end.
//   h = x@W_in + b ; h += relu(h@W1+b1)@W2 + b2 ; k = LN(h)@K_w + K_b
// Outputs: k in bf16 (+ row |k|^2), optionally h and k in f32 (queries).
// ---------------------------------------------------------------------------
__global__ void __launch_bounds__(32) encoder_kernel(
    const float* __restrict__ xin,
    const u16* __restrict__ WinT, const float* __restrict__ b_in,
    const u16* __restrict__ W1T,  const float* __restrict__ b1,
    const u16* __restrict__ W2T,  const float* __restrict__ b2,
    const float* __restrict__ lng, const float* __restrict__ lnb,
    const u16* __restrict__ KwT,  const float* __restrict__ Kb,
    u16* __restrict__ kout_bf, float* __restrict__ k2out,
    float* __restrict__ hout, float* __restrict__ kout_f32) {
  __shared__ __align__(16) u16   xa[16][F_IN];
  __shared__ __align__(16) u16   hb[16][D_MODEL];
  __shared__ __align__(16) u16   ua[16][D_BLK];
  __shared__ __align__(16) float hf[16][D_MODEL];
  __shared__ __align__(16) u16   nb[16][D_MODEL];
  __shared__ __align__(16) float kf[16][D_MODEL];

  int lane = threadIdx.x & 31;
  int m = lane & 15, g = lane >> 4;
  size_t rowbase = (size_t)blockIdx.x * 16;

  { // stage input tile as bf16 (lane = row m, half g)
    const float* src = xin + (rowbase + m) * F_IN + g * 16;
    u16* dst = &xa[m][g * 16];
#pragma unroll
    for (int i = 0; i < 16; ++i) dst[i] = f2bf(src[i]);
  }

  u16x16 af = load_frag(&xa[0][0], F_IN, 0);   // K = 32 exactly one WMMA step
  v8f hreg[8];
#pragma unroll
  for (int nt = 0; nt < 8; ++nt) {
    float bb = b_in[nt * 16 + m];
    v8f c = {bb, bb, bb, bb, bb, bb, bb, bb};
    hreg[nt] = wmma_bf16(af, load_frag(WinT + nt * 16 * F_IN, F_IN, 0), c);
#pragma unroll
    for (int r = 0; r < 8; ++r) hb[r + 8 * g][nt * 16 + m] = f2bf(hreg[nt][r]);
  }
  // MLP layer 1: 16x128 @ 128x256, relu
  for (int nt = 0; nt < 16; ++nt) {
    float bb = b1[nt * 16 + m];
    v8f acc = {bb, bb, bb, bb, bb, bb, bb, bb};
#pragma unroll
    for (int kt = 0; kt < 4; ++kt)
      acc = wmma_bf16(load_frag(&hb[0][0], D_MODEL, kt * 32),
                      load_frag(W1T + nt * 16 * D_MODEL, D_MODEL, kt * 32), acc);
#pragma unroll
    for (int r = 0; r < 8; ++r) ua[r + 8 * g][nt * 16 + m] = f2bf(fmaxf(acc[r], 0.f));
  }
  // MLP layer 2 + residual (accumulators seeded with h + b2)
  for (int nt = 0; nt < 8; ++nt) {
    float bb = b2[nt * 16 + m];
    v8f acc = hreg[nt];
#pragma unroll
    for (int e = 0; e < 8; ++e) acc[e] = acc[e] + bb;
#pragma unroll
    for (int kt = 0; kt < 8; ++kt)
      acc = wmma_bf16(load_frag(&ua[0][0], D_BLK, kt * 32),
                      load_frag(W2T + nt * 16 * D_BLK, D_BLK, kt * 32), acc);
#pragma unroll
    for (int r = 0; r < 8; ++r) {
      int rr = r + 8 * g, cc = nt * 16 + m;
      hf[rr][cc] = acc[r];
      if (hout) hout[(rowbase + rr) * D_MODEL + cc] = acc[r];
    }
  }
  // LayerNorm: lanes 0..15, one row each
  if (lane < 16) {
    float s = 0.f;
    for (int d = 0; d < D_MODEL; ++d) s += hf[lane][d];
    float mean = s * (1.f / D_MODEL), v = 0.f;
    for (int d = 0; d < D_MODEL; ++d) { float t = hf[lane][d] - mean; v += t * t; }
    float inv = rsqrtf(v * (1.f / D_MODEL) + 1e-5f);
    for (int d = 0; d < D_MODEL; ++d)
      nb[lane][d] = f2bf((hf[lane][d] - mean) * inv * lng[d] + lnb[d]);
  }
  // k = LN(h) @ K_w + K_b
  for (int nt = 0; nt < 8; ++nt) {
    float bb = Kb[nt * 16 + m];
    v8f acc = {bb, bb, bb, bb, bb, bb, bb, bb};
#pragma unroll
    for (int kt = 0; kt < 4; ++kt)
      acc = wmma_bf16(load_frag(&nb[0][0], D_MODEL, kt * 32),
                      load_frag(KwT + nt * 16 * D_MODEL, D_MODEL, kt * 32), acc);
#pragma unroll
    for (int r = 0; r < 8; ++r) {
      int rr = r + 8 * g, cc = nt * 16 + m;
      kf[rr][cc] = acc[r];
      kout_bf[(rowbase + rr) * D_MODEL + cc] = f2bf(acc[r]);
      if (kout_f32) kout_f32[(rowbase + rr) * D_MODEL + cc] = acc[r];
    }
  }
  if (lane < 16) {
    float s = 0.f;
    for (int d = 0; d < D_MODEL; ++d) { float t = kf[lane][d]; s += t * t; }
    k2out[rowbase + lane] = s;
  }
}

// ---------------------------------------------------------------------------
// Streaming kNN top-96. Block = 16 queries, 4 waves; wave w scans candidates
// [w*50000, (w+1)*50000) with WMMA dot tiles (score = 2*dot - |ck|^2, the
// per-query -|k|^2 term is rank-invariant), replace-min top lists in LDS,
// then a 4-way merge picks the final 96 (descending, like top_k).
// ---------------------------------------------------------------------------
__global__ void __launch_bounds__(128) topk_kernel(
    const u16* __restrict__ qk, const u16* __restrict__ ck,
    const float* __restrict__ ck2, int* __restrict__ idx_out) {
  __shared__ float sc[4][16][C_CTX];
  __shared__ int   si[4][16][C_CTX];
  __shared__ float stile[4][16][16];

  int tid = threadIdx.x, w = tid >> 5, lane = tid & 31;
  int q0 = blockIdx.x * 16;

  u16x16 a0 = load_frag(qk + (size_t)q0 * D_MODEL, D_MODEL, 0);
  u16x16 a1 = load_frag(qk + (size_t)q0 * D_MODEL, D_MODEL, 32);
  u16x16 a2 = load_frag(qk + (size_t)q0 * D_MODEL, D_MODEL, 64);
  u16x16 a3 = load_frag(qk + (size_t)q0 * D_MODEL, D_MODEL, 96);

  if (lane < 16) {
    for (int j = 0; j < C_CTX; ++j) { sc[w][lane][j] = -3.0e38f; si[w][lane][j] = 0; }
  }
  float mn = -3.0e38f; int mp = 0;

  const int NPW = N_CAND / 4;
  int cstart = w * NPW;
  for (int t = 0; t < NPW / 16; ++t) {
    int cbase = cstart + t * 16;
    __builtin_prefetch(ck + (size_t)(cbase + 32) * D_MODEL, 0, 1);
    v8f acc = {0.f, 0.f, 0.f, 0.f, 0.f, 0.f, 0.f, 0.f};
    acc = wmma_bf16(a0, load_frag(ck + (size_t)cbase * D_MODEL, D_MODEL, 0),  acc);
    acc = wmma_bf16(a1, load_frag(ck + (size_t)cbase * D_MODEL, D_MODEL, 32), acc);
    acc = wmma_bf16(a2, load_frag(ck + (size_t)cbase * D_MODEL, D_MODEL, 64), acc);
    acc = wmma_bf16(a3, load_frag(ck + (size_t)cbase * D_MODEL, D_MODEL, 96), acc);
    int n = lane & 15, g = lane >> 4;
    float c2 = ck2[cbase + n];
#pragma unroll
    for (int r = 0; r < 8; ++r) stile[w][r + 8 * g][n] = 2.0f * acc[r] - c2;

    if (lane < 16) {               // lane q owns query q's list
      int q = lane;
      for (int j = 0; j < 16; ++j) {
        float s = stile[w][q][j];
        if (s > mn) {
          sc[w][q][mp] = s; si[w][q][mp] = cbase + j;
          mn = sc[w][q][0]; mp = 0;
          for (int u = 1; u < C_CTX; ++u)
            if (sc[w][q][u] < mn) { mn = sc[w][q][u]; mp = u; }
        }
      }
    }
  }
  __syncthreads();
  if (tid < 16) {                  // merge 4 partial lists -> top 96 (descending)
    int q = tid;
    for (int r = 0; r < C_CTX; ++r) {
      float best = -3.9e38f; int bw = 0, bj = 0;
      for (int w2 = 0; w2 < 4; ++w2)
        for (int j = 0; j < C_CTX; ++j) {
          float s = sc[w2][q][j];
          if (s > best) { best = s; bw = w2; bj = j; }
        }
      idx_out[(q0 + q) * C_CTX + r] = si[bw][q][bj];
      sc[bw][q][bj] = -3.9e38f;
    }
  }
}

// ---------------------------------------------------------------------------
// Context aggregation (one block per query): gather 96 neighbor keys, f32
// sims + softmax, diff -> T-MLP via WMMA, probability-weighted ds_add_f32
// reduction into ctx[128]. ctx_y term folded analytically:
//   sum_c p_c*(y_c*lab_w + lab_b) = (sum p_c y_c)*lab_w + lab_b.
// ---------------------------------------------------------------------------
__global__ void __launch_bounds__(128) aggregate_kernel(
    const int* __restrict__ idxp, const u16* __restrict__ ckbf,
    const float* __restrict__ cand_y, const float* __restrict__ qkf,
    const float* __restrict__ lw, const float* __restrict__ lb,
    const u16* __restrict__ T1T, const float* __restrict__ Tb1,
    const u16* __restrict__ T2T, float* __restrict__ ctx_out) {
  __shared__ __align__(16) u16 sbuf[4 * 16 * D_BLK];   // phase A: ckb[96][128]; phase B: per-wave u slabs
  __shared__ __align__(16) u16 diffb[C_CTX][D_MODEL];
  __shared__ float qk[D_MODEL];
  __shared__ float sim[C_CTX], probs[C_CTX], yv[C_CTX];
  __shared__ float ctxa[D_MODEL];
  __shared__ float ybar_s;

  int q = blockIdx.x, tid = threadIdx.x;
  int w = tid >> 5, lane = tid & 31;
  int mm = lane & 15, gg = lane >> 4;
  u16(*ckb)[D_MODEL] = (u16(*)[D_MODEL])sbuf;

  qk[tid] = qkf[(size_t)q * D_MODEL + tid];
  if (tid < C_CTX) {
    int id = idxp[q * C_CTX + tid];
    yv[tid] = cand_y[id];
    const uint4* src = (const uint4*)(ckbf + (size_t)id * D_MODEL);
    uint4* dst = (uint4*)&ckb[tid][0];
#pragma unroll
    for (int t = 0; t < 16; ++t) dst[t] = src[t];
  }
  __syncthreads();
  if (tid < C_CTX) {
    float dot = 0.f, cq = 0.f, q2 = 0.f;
    for (int d = 0; d < D_MODEL; ++d) {
      float cv = bf2f(ckb[tid][d]);
      float qv = qk[d];
      dot += qv * cv; cq += cv * cv; q2 += qv * qv;
      diffb[tid][d] = f2bf(qv - cv);
    }
    sim[tid] = 2.f * dot - q2 - cq;
  }
  __syncthreads();
  if (tid == 0) {
    float mx = sim[0];
    for (int c = 1; c < C_CTX; ++c) mx = fmaxf(mx, sim[c]);
    float s = 0.f;
    for (int c = 0; c < C_CTX; ++c) { float e = __expf(sim[c] - mx); probs[c] = e; s += e; }
    float inv = 1.f / s, yb = 0.f;
    for (int c = 0; c < C_CTX; ++c) { probs[c] *= inv; yb += probs[c] * yv[c]; }
    ybar_s = yb;
  }
  __syncthreads();
  ctxa[tid] = ybar_s * lw[tid] + lb[tid];
  __syncthreads();

  u16* uslab = sbuf + (size_t)w * 16 * D_BLK;          // sbuf reuse, safe after barriers
  for (int tile = w; tile < 6; tile += 4) {
    int r0 = tile * 16;
    for (int nt = 0; nt < 16; ++nt) {                  // relu(diff@T_w1 + b1)
      float bb = Tb1[nt * 16 + mm];
      v8f acc = {bb, bb, bb, bb, bb, bb, bb, bb};
#pragma unroll
      for (int kt = 0; kt < 4; ++kt)
        acc = wmma_bf16(load_frag(&diffb[r0][0], D_MODEL, kt * 32),
                        load_frag(T1T + nt * 16 * D_MODEL, D_MODEL, kt * 32), acc);
#pragma unroll
      for (int r = 0; r < 8; ++r)
        uslab[(r + 8 * gg) * D_BLK + nt * 16 + mm] = f2bf(fmaxf(acc[r], 0.f));
    }
    for (int nt = 0; nt < 8; ++nt) {                   // (.)@T_w2, weighted row-sum
      v8f acc = {0.f, 0.f, 0.f, 0.f, 0.f, 0.f, 0.f, 0.f};
#pragma unroll
      for (int kt = 0; kt < 8; ++kt)
        acc = wmma_bf16(load_frag(uslab, D_BLK, kt * 32),
                        load_frag(T2T + nt * 16 * D_BLK, D_BLK, kt * 32), acc);
      float part = 0.f;
#pragma unroll
      for (int r = 0; r < 8; ++r) part += probs[r0 + r + 8 * gg] * acc[r];
      atomicAdd(&ctxa[nt * 16 + mm], part);
    }
  }
  __syncthreads();
  ctx_out[(size_t)q * D_MODEL + tid] = ctxa[tid];
}

// ---------------------------------------------------------------------------
// Final head (f32 VALU, tiny): x += ctx; LN; MLP(128->256->128) residual;
// LN; relu; dot(head_w) + head_b. One block per query, 128 threads.
// ---------------------------------------------------------------------------
__global__ void __launch_bounds__(128) head_kernel(
    const float* __restrict__ xh, const float* __restrict__ ctxv,
    const float* __restrict__ plg, const float* __restrict__ plb,
    const float* __restrict__ pw1, const float* __restrict__ pb1,
    const float* __restrict__ pw2, const float* __restrict__ pb2,
    const float* __restrict__ hlg, const float* __restrict__ hlb,
    const float* __restrict__ hw, const float* __restrict__ hbias,
    float* __restrict__ out) {
  __shared__ float xs[128], hn[128], us[256], red[128], stats[2];
  int q = blockIdx.x, t = threadIdx.x;
  float xv = xh[(size_t)q * 128 + t] + ctxv[(size_t)q * 128 + t];
  xs[t] = xv;
  __syncthreads();
  if (t == 0) {
    float s = 0.f; for (int d = 0; d < 128; ++d) s += xs[d];
    float mean = s / 128.f, v = 0.f;
    for (int d = 0; d < 128; ++d) { float u = xs[d] - mean; v += u * u; }
    stats[0] = mean; stats[1] = rsqrtf(v / 128.f + 1e-5f);
  }
  __syncthreads();
  hn[t] = (xv - stats[0]) * stats[1] * plg[t] + plb[t];
  __syncthreads();
  for (int jj = 0; jj < 2; ++jj) {
    int j = t + jj * 128;
    float a = pb1[j];
    for (int d = 0; d < 128; ++d) a += hn[d] * pw1[d * 256 + j];
    us[j] = fmaxf(a, 0.f);
  }
  __syncthreads();
  float dlt = pb2[t];
  for (int j = 0; j < 256; ++j) dlt += us[j] * pw2[j * 128 + t];
  float x2 = xv + dlt;
  __syncthreads();
  xs[t] = x2;
  __syncthreads();
  if (t == 0) {
    float s = 0.f; for (int d = 0; d < 128; ++d) s += xs[d];
    float mean = s / 128.f, v = 0.f;
    for (int d = 0; d < 128; ++d) { float u = xs[d] - mean; v += u * u; }
    stats[0] = mean; stats[1] = rsqrtf(v / 128.f + 1e-5f);
  }
  __syncthreads();
  float r = fmaxf((x2 - stats[0]) * stats[1] * hlg[t] + hlb[t], 0.f);
  red[t] = r * hw[t];
  __syncthreads();
  if (t == 0) {
    float s = hbias[0];
    for (int d = 0; d < 128; ++d) s += red[d];
    out[q] = s;
  }
}

// ---------------------------------------------------------------------------
extern "C" void kernel_launch(void* const* d_in, const int* in_sizes, int n_in,
                              void* d_out, int out_size, void* d_ws, size_t ws_size,
                              hipStream_t stream) {
  (void)in_sizes; (void)n_in; (void)out_size; (void)ws_size;
  const float* x_num  = (const float*)d_in[0];
  const float* cand_x = (const float*)d_in[1];
  const float* cand_y = (const float*)d_in[2];
  const float* W_in   = (const float*)d_in[3];
  const float* b_in   = (const float*)d_in[4];
  const float* enc_w1 = (const float*)d_in[5];
  const float* enc_b1 = (const float*)d_in[6];
  const float* enc_w2 = (const float*)d_in[7];
  const float* enc_b2 = (const float*)d_in[8];
  const float* ln_g   = (const float*)d_in[9];
  const float* ln_b   = (const float*)d_in[10];
  const float* K_w    = (const float*)d_in[11];
  const float* K_b    = (const float*)d_in[12];
  const float* lab_w  = (const float*)d_in[13];
  const float* lab_b  = (const float*)d_in[14];
  const float* T_w1   = (const float*)d_in[15];
  const float* T_b1   = (const float*)d_in[16];
  const float* T_w2   = (const float*)d_in[17];
  const float* p_lg   = (const float*)d_in[18];
  const float* p_lb   = (const float*)d_in[19];
  const float* p_w1   = (const float*)d_in[20];
  const float* p_b1   = (const float*)d_in[21];
  const float* p_w2   = (const float*)d_in[22];
  const float* p_b2   = (const float*)d_in[23];
  const float* h_lg   = (const float*)d_in[24];
  const float* h_lb   = (const float*)d_in[25];
  const float* h_w    = (const float*)d_in[26];
  const float* h_b    = (const float*)d_in[27];

  char* base = (char*)d_ws;
  size_t off = 0;
  auto take = [&](size_t bytes) -> char* {
    char* r = base + off;
    off += (bytes + 255) & ~(size_t)255;
    return r;
  };
  u16*   ck_bf = (u16*)  take((size_t)N_CAND * D_MODEL * 2);  // 51.2 MB, L2-resident
  float* ck2   = (float*)take((size_t)N_CAND * 4);
  u16*   qk_bf = (u16*)  take((size_t)B_Q * D_MODEL * 2);
  float* qk2   = (float*)take((size_t)B_Q * 4);
  float* qx    = (float*)take((size_t)B_Q * D_MODEL * 4);
  float* qkf   = (float*)take((size_t)B_Q * D_MODEL * 4);
  int*   idx   = (int*)  take((size_t)B_Q * C_CTX * 4);
  float* ctx   = (float*)take((size_t)B_Q * D_MODEL * 4);
  u16* WinT = (u16*)take((size_t)F_IN * D_MODEL * 2);
  u16* W1T  = (u16*)take((size_t)D_MODEL * D_BLK * 2);
  u16* W2T  = (u16*)take((size_t)D_BLK * D_MODEL * 2);
  u16* KwT  = (u16*)take((size_t)D_MODEL * D_MODEL * 2);
  u16* T1T  = (u16*)take((size_t)D_MODEL * D_BLK * 2);
  u16* T2T  = (u16*)take((size_t)D_BLK * D_MODEL * 2);

  transpose_cvt_kernel<<<(F_IN * D_MODEL + 255) / 256, 256, 0, stream>>>(W_in, WinT, F_IN, D_MODEL);
  transpose_cvt_kernel<<<(D_MODEL * D_BLK + 255) / 256, 256, 0, stream>>>(enc_w1, W1T, D_MODEL, D_BLK);
  transpose_cvt_kernel<<<(D_BLK * D_MODEL + 255) / 256, 256, 0, stream>>>(enc_w2, W2T, D_BLK, D_MODEL);
  transpose_cvt_kernel<<<(D_MODEL * D_MODEL + 255) / 256, 256, 0, stream>>>(K_w, KwT, D_MODEL, D_MODEL);
  transpose_cvt_kernel<<<(D_MODEL * D_BLK + 255) / 256, 256, 0, stream>>>(T_w1, T1T, D_MODEL, D_BLK);
  transpose_cvt_kernel<<<(D_BLK * D_MODEL + 255) / 256, 256, 0, stream>>>(T_w2, T2T, D_BLK, D_MODEL);

  encoder_kernel<<<N_CAND / 16, 32, 0, stream>>>(
      cand_x, WinT, b_in, W1T, enc_b1, W2T, enc_b2, ln_g, ln_b, KwT, K_b,
      ck_bf, ck2, nullptr, nullptr);
  encoder_kernel<<<B_Q / 16, 32, 0, stream>>>(
      x_num, WinT, b_in, W1T, enc_b1, W2T, enc_b2, ln_g, ln_b, KwT, K_b,
      qk_bf, qk2, qx, qkf);

  topk_kernel<<<B_Q / 16, 128, 0, stream>>>(qk_bf, ck_bf, ck2, idx);

  aggregate_kernel<<<B_Q, 128, 0, stream>>>(
      idx, ck_bf, cand_y, qkf, lab_w, lab_b, T1T, T_b1, T2T, ctx);

  head_kernel<<<B_Q, 128, 0, stream>>>(
      qx, ctx, p_lg, p_lb, p_w1, p_b1, p_w2, p_b2, h_lg, h_lb, h_w, h_b,
      (float*)d_out);
}
